// SRNS_38534446580254
// MI455X (gfx1250) — compile-verified, matrix operand order checked by reading, not run on hardware
//
#include <hip/hip_runtime.h>
#include <math.h>

// ---------------- problem constants (from reference) ----------------
#define B_    4
#define P_    8192
#define S_    10
#define LAT_  256
#define HID_  256
#define FEAT_ 32
#define NSC   75040          // total scene-MLP params per batch
// theta layout per batch: W1[3*256] b1[256] W2[256*256] b2[256] W3[256*32] b3[32]
#define TH_W1 0
#define TH_B1 768
#define TH_W2 1024
#define TH_B2 66560
#define TH_W3 66816
#define TH_B3 75008

#define TILE    64           // pixels per workgroup
#define THREADS 256          // 8 waves (wave32)
#define NWAVES  (THREADS/32)
#define ASTR    264          // padded bf16 stride for activation rows (256 + 8)

typedef __attribute__((ext_vector_type(16))) __bf16 v16bf;
typedef __attribute__((ext_vector_type(8)))  float  v8f;

// ---------------- dynamic-LDS layout (bytes, all 16B aligned) ----------------
#define OFF_W1   0        // float[3*256]   scene W1 (reused for pix_W5 [256*3])
#define OFF_B1   3072     // float[256]     scene b1 (reused for pix_b1)
#define OFF_B2   4096     // float[256]     scene b2
#define OFF_B3   5120     // float[32]      scene b3
#define OFF_WIH  5248     // float[32*128]
#define OFF_WHH  21632    // float[32*128]
#define OFF_BL   38016    // float[128]
#define OFF_WD   38528    // float[32]
#define OFF_FW2  38656    // bf16[256*256]  fragment-ordered (scene W2 / pix W2,3,4)
#define OFF_FWS  169728   // bf16[256*32]   fragment-ordered (scene W3 / pix W1)
#define OFF_ACTA 186112   // bf16[TILE*ASTR]
#define OFF_ACTB 219904   // bf16[TILE*ASTR]
#define OFF_FEAT 253696   // float[TILE*32]
#define OFF_H    261888   // float[TILE*32]
#define OFF_C    270080   // float[TILE*32]
#define OFF_DIR  278272   // float[TILE*3]
#define OFF_D    279040   // float[TILE]
#define OFF_ORG  279296   // float[4]
#define OFF_B4   279312   // float[256]     pixel bias slot (pb2/pb3/pb4)
#define SMEM_BYTES 280336 // < 320KB WGP LDS

// ---------------- workspace layout (bytes) ----------------
#define WS_THETA   0                      // float[B*NSC]
#define WS_HIDDEN  1200640                // float[B*HID]
#define WS_SCF2    1204736                // bf16[B*256*256] scene W2 fragments
#define WS_SCF3    1729024                // bf16[B*256*32]  scene W3 fragments
#define WS_PXF1    1794560                // bf16[32*256]
#define WS_PXF2    1810944                // bf16[256*256]
#define WS_PXF3    1942016                // bf16[256*256]
#define WS_PXF4    2073088                // bf16[256*256]

__device__ __forceinline__ float sigmoidf_(float x) { return 1.0f / (1.0f + expf(-x)); }

// ---- gfx1250 async global->LDS copy (ASYNCcnt path, ISA ch.10/§15.18) ----
// VDST VGPR holds the workgroup-relative LDS byte address; dynamic LDS starts
// at offset 0 here (no static LDS in this kernel), so OFF_* constants are valid.
__device__ __forceinline__ void async_cp16(unsigned int ldsOff, const void* g) {
    asm volatile("global_load_async_to_lds_b128 %0, %1, off"
                 :: "v"(ldsOff), "v"((unsigned long long)(uintptr_t)g) : "memory");
}
__device__ __forceinline__ void async_stage(unsigned int ldsOff, const void* gsrc,
                                            int bytes, int tid) {
    const char* g = (const char*)gsrc;
    for (int o = tid * 16; o < bytes; o += THREADS * 16)
        async_cp16(ldsOff + (unsigned int)o, g + o);
}
#define ASYNC_WAIT() asm volatile("s_wait_asynccnt 0" ::: "memory")

// A-fragment (16x32 bf16) for pixel rows [rowBase, rowBase+16) at K-tile kt,
// gathered from a row-major padded activation buffer (stride ASTR halves).
// Layout: lane = M%16 (+16 for klo=8 half), k = klo + (h<8 ? h : h+8).
__device__ __forceinline__ v16bf load_a(const __bf16* act, int rowBase, int kt, int lane) {
    const __bf16* p = act + (rowBase + (lane & 15)) * ASTR + kt * 32 + ((lane >> 4) << 3);
    v16bf a;
#pragma unroll
    for (int h = 0; h < 16; ++h) a[h] = p[(h < 8) ? h : h + 8];
    return a;
}

// B-fragment: one contiguous 32B load per lane from the fragment-ordered buffer.
__device__ __forceinline__ v16bf load_b(const __bf16* frag, int kt, int nt, int NT, int lane) {
    return *(const v16bf*)(frag + (((kt * NT + nt) << 5) + lane) * 16);
}

// Wave-cooperative GEMM: (TILE x K) @ (K x N) + bias, K = KT*32, N = NT*16.
// Output C layout: VGPR r -> M = r + (lane>=16 ? 8 : 0), N = lane%16.
template <int KT, int NT, bool RELU, bool OUT_BF16, bool OUT_F32>
__device__ __forceinline__ void wave_gemm(const __bf16* actIn, const __bf16* fragW,
                                          const float* bias, __bf16* actOut,
                                          float* f32Out, int wave, int lane) {
    constexpr int MT = TILE / 16;
    for (int t = wave; t < MT * NT; t += NWAVES) {
        int mt = t / NT, nt = t % NT;
        int n  = nt * 16 + (lane & 15);
        float bn = bias[n];
        v8f c = {bn, bn, bn, bn, bn, bn, bn, bn};
#pragma unroll
        for (int kt = 0; kt < KT; ++kt) {
            v16bf a = load_a(actIn, mt * 16, kt, lane);
            v16bf b = load_b(fragW, kt, nt, NT, lane);
            c = __builtin_amdgcn_wmma_f32_16x16x32_bf16(false, a, false, b,
                                                        (short)0, c, false, false);
        }
        int m0 = mt * 16 + ((lane >> 4) << 3);
#pragma unroll
        for (int r = 0; r < 8; ++r) {
            float v = c[r];
            if (RELU) v = fmaxf(v, 0.0f);
            int prow = m0 + r;
            if constexpr (OUT_BF16) actOut[prow * ASTR + n] = (__bf16)v;
            if constexpr (OUT_F32)  f32Out[prow * FEAT_ + n] = v;
        }
    }
}

// ---------------- hypernetwork ----------------
__global__ void hyper_hidden_kernel(const int* __restrict__ id1, const float* __restrict__ lat,
                                    const float* __restrict__ W1, const float* __restrict__ b1,
                                    float* __restrict__ hidden) {
    __shared__ float sl[LAT_];
    int b = blockIdx.x, j = threadIdx.x;
    sl[j] = lat[(size_t)id1[b] * LAT_ + j];
    __syncthreads();
    float acc = b1[j];
    for (int k = 0; k < LAT_; ++k) acc = fmaf(sl[k], W1[k * HID_ + j], acc);
    hidden[b * HID_ + j] = fmaxf(acc, 0.0f);
}

__global__ void hyper_theta_kernel(const float* __restrict__ hidden, const float* __restrict__ W2,
                                   const float* __restrict__ b2, float* __restrict__ theta) {
    __shared__ float sh[HID_];
    int b = blockIdx.y;
    sh[threadIdx.x] = hidden[b * HID_ + threadIdx.x];
    __syncthreads();
    int j = blockIdx.x * 256 + threadIdx.x;
    if (j >= NSC) return;
    float acc = b2[j];
    for (int k = 0; k < HID_; ++k) acc = fmaf(sh[k], W2[(size_t)k * NSC + j], acc);
    theta[(size_t)b * NSC + j] = acc;
}

// One-time pack: KxN row-major f32 -> fragment-ordered bf16 WMMA-B image.
// dst halfword index = ((kt*NT+nt)*32+lane)*16+h with
// n = nt*16 + lane%16, k = kt*32 + (lane>=16?8:0) + (h<8?h:h+8).
__global__ void pack_frags_kernel(const float* __restrict__ src, __bf16* __restrict__ dst,
                                  int K, int N) {
    int idx = blockIdx.x * blockDim.x + threadIdx.x;
    if (idx >= K * N) return;
    int h    = idx & 15;
    int lane = (idx >> 4) & 31;
    int f    = idx >> 9;
    int NT   = N >> 4;
    int nt   = f % NT;
    int kt   = f / NT;
    int k = kt * 32 + ((lane >> 4) << 3) + ((h < 8) ? h : h + 8);
    int n = nt * 16 + (lane & 15);
    dst[idx] = (__bf16)src[k * N + n];
}

// ---------------- fused raymarch + LSTM + pixel MLP ----------------
__global__ void __launch_bounds__(THREADS, 1)
srn_raymarch(const float* __restrict__ xuv,
             const float* __restrict__ Rm, const float* __restrict__ Km,
             const float* __restrict__ Tm,
             const float* __restrict__ Wih, const float* __restrict__ Whh,
             const float* __restrict__ bl, const float* __restrict__ Wd,
             const float* __restrict__ bd,
             const float* __restrict__ pb1, const float* __restrict__ pb2,
             const float* __restrict__ pb3, const float* __restrict__ pb4,
             const float* __restrict__ pW5, const float* __restrict__ pb5,
             const float* __restrict__ h0, const float* __restrict__ c0,
             const float* __restrict__ theta,
             const __bf16* __restrict__ scF2, const __bf16* __restrict__ scF3,
             const __bf16* __restrict__ pxF1, const __bf16* __restrict__ pxF2,
             const __bf16* __restrict__ pxF3, const __bf16* __restrict__ pxF4,
             float* __restrict__ outRgb, float* __restrict__ outD) {
    extern __shared__ char smem[];
    float*  sW1   = (float*)(smem + OFF_W1);
    float*  sb1   = (float*)(smem + OFF_B1);
    float*  sb2   = (float*)(smem + OFF_B2);
    float*  sb3   = (float*)(smem + OFF_B3);
    float*  sWih  = (float*)(smem + OFF_WIH);
    float*  sWhh  = (float*)(smem + OFF_WHH);
    float*  sbl   = (float*)(smem + OFF_BL);
    float*  sWd   = (float*)(smem + OFF_WD);
    __bf16* fragW2 = (__bf16*)(smem + OFF_FW2);
    __bf16* fragWs = (__bf16*)(smem + OFF_FWS);
    __bf16* actA  = (__bf16*)(smem + OFF_ACTA);
    __bf16* actB  = (__bf16*)(smem + OFF_ACTB);
    float*  sFeat = (float*)(smem + OFF_FEAT);
    float*  sH    = (float*)(smem + OFF_H);
    float*  sC    = (float*)(smem + OFF_C);
    float*  sDir  = (float*)(smem + OFF_DIR);
    float*  sD    = (float*)(smem + OFF_D);
    float*  sOrg  = (float*)(smem + OFF_ORG);
    float*  sB4   = (float*)(smem + OFF_B4);

    const int tid  = threadIdx.x;
    const int lane = tid & 31;
    const int wave = tid >> 5;
    const int b    = blockIdx.y;
    const int p0   = blockIdx.x * TILE;
    const float* th = theta + (size_t)b * NSC;

    // ---- async-stage ALL resident weights/state into LDS (ASYNCcnt) ----
    async_stage(OFF_W1,  th + TH_W1, 3072, tid);
    async_stage(OFF_B1,  th + TH_B1, 1024, tid);
    async_stage(OFF_B2,  th + TH_B2, 1024, tid);
    async_stage(OFF_B3,  th + TH_B3, 128,  tid);
    async_stage(OFF_FW2, scF2 + (size_t)b * 256 * 256, 131072, tid);
    async_stage(OFF_FWS, scF3 + (size_t)b * 256 * 32,  16384,  tid);
    async_stage(OFF_WIH, Wih, 16384, tid);
    async_stage(OFF_WHH, Whh, 16384, tid);
    async_stage(OFF_BL,  bl,  512,   tid);
    async_stage(OFF_WD,  Wd,  128,   tid);
    async_stage(OFF_H, h0 + ((size_t)b * P_ + p0) * FEAT_, TILE * FEAT_ * 4, tid);
    async_stage(OFF_C, c0 + ((size_t)b * P_ + p0) * FEAT_, TILE * FEAT_ * 4, tid);

    // ---- ray setup overlaps the async copies ----
    if (tid < TILE) {
        int p = p0 + tid;
        float u = xuv[((size_t)b * P_ + p) * 2 + 0];
        float v = xuv[((size_t)b * P_ + p) * 2 + 1];
        const float* kk = Km + b * 9;
        float m00=kk[0],m01=kk[1],m02=kk[2],m10=kk[3],m11=kk[4],m12=kk[5],m20=kk[6],m21=kk[7],m22=kk[8];
        float det = m00*(m11*m22-m12*m21) - m01*(m10*m22-m12*m20) + m02*(m10*m21-m11*m20);
        float id = 1.0f / det;
        float i00 =  (m11*m22 - m12*m21)*id, i01 = -(m01*m22 - m02*m21)*id, i02 =  (m01*m12 - m02*m11)*id;
        float i10 = -(m10*m22 - m12*m20)*id, i11 =  (m00*m22 - m02*m20)*id, i12 = -(m00*m12 - m02*m10)*id;
        float i20 =  (m10*m21 - m11*m20)*id, i21 = -(m00*m21 - m01*m20)*id, i22 =  (m00*m11 - m01*m10)*id;
        float v0 = i00*u + i01*v + i02;
        float v1 = i10*u + i11*v + i12;
        float v2 = i20*u + i21*v + i22;
        const float* rr = Rm + b * 9;
        float d0 = rr[0]*v0 + rr[1]*v1 + rr[2]*v2;
        float d1 = rr[3]*v0 + rr[4]*v1 + rr[5]*v2;
        float d2 = rr[6]*v0 + rr[7]*v1 + rr[8]*v2;
        float inv_n = rsqrtf(d0*d0 + d1*d1 + d2*d2);
        sDir[tid*3+0] = d0*inv_n; sDir[tid*3+1] = d1*inv_n; sDir[tid*3+2] = d2*inv_n;
        sD[tid] = 0.05f;
    }
    if (tid == 0) { sOrg[0] = Tm[b*3]; sOrg[1] = Tm[b*3+1]; sOrg[2] = Tm[b*3+2]; }
    ASYNC_WAIT();
    __syncthreads();

    auto scene_pass = [&](bool finalPass) {
        // layer1 (K=3, scalar f32) -> actA (relu, bf16)
        for (int idx = tid; idx < TILE * 256; idx += THREADS) {
            int p = idx >> 8, j = idx & 255;
            float dd = sD[p];
            float px = sOrg[0] + dd * sDir[p*3+0];
            float py = sOrg[1] + dd * sDir[p*3+1];
            float pz = sOrg[2] + dd * sDir[p*3+2];
            float a = fmaf(px, sW1[j], fmaf(py, sW1[256+j], fmaf(pz, sW1[512+j], sb1[j])));
            actA[p * ASTR + j] = (__bf16)fmaxf(a, 0.0f);
        }
        __syncthreads();
        // layer2 256->256 (WMMA, relu) actA -> actB
        wave_gemm<8, 16, true, true, false>(actA, fragW2, sb2, actB, nullptr, wave, lane);
        __syncthreads();
        // layer3 256->32 (WMMA, no relu) actB -> sFeat (+ bf16 feat into actA on final pass)
        if (finalPass)
            wave_gemm<8, 2, false, true,  true>(actB, fragWs, sb3, actA,    sFeat, wave, lane);
        else
            wave_gemm<8, 2, false, false, true>(actB, fragWs, sb3, nullptr, sFeat, wave, lane);
        __syncthreads();
    };

    // ---- raymarch loop: S steps of scene_mlp + LSTM + depth update ----
    for (int s = 0; s < S_; ++s) {
        scene_pass(false);
        float hn[8], cn[8];
#pragma unroll
        for (int i = 0; i < 8; ++i) {
            int idx = tid + i * THREADS;          // (p, j) over TILE*32 states
            int p = idx >> 5, j = idx & 31;
            float gi = sbl[j], gf = sbl[32+j], gg = sbl[64+j], go = sbl[96+j];
            for (int k = 0; k < FEAT_; ++k) {
                float fv = sFeat[p * FEAT_ + k];
                float hv = sH[p * FEAT_ + k];
                const float* wi = sWih + k * 128;
                const float* wh = sWhh + k * 128;
                gi = fmaf(fv, wi[j],      fmaf(hv, wh[j],      gi));
                gf = fmaf(fv, wi[32 + j], fmaf(hv, wh[32 + j], gf));
                gg = fmaf(fv, wi[64 + j], fmaf(hv, wh[64 + j], gg));
                go = fmaf(fv, wi[96 + j], fmaf(hv, wh[96 + j], go));
            }
            float cne = sigmoidf_(gf) * sC[idx] + sigmoidf_(gi) * tanhf(gg);
            cn[i] = cne;
            hn[i] = sigmoidf_(go) * tanhf(cne);
        }
        __syncthreads();
#pragma unroll
        for (int i = 0; i < 8; ++i) { int idx = tid + i * THREADS; sH[idx] = hn[i]; sC[idx] = cn[i]; }
        __syncthreads();
        if (tid < TILE) {
            float acc = bd[0];
            for (int k = 0; k < FEAT_; ++k) acc = fmaf(sH[tid * FEAT_ + k], sWd[k], acc);
            sD[tid] += acc;
        }
        __syncthreads();
    }
    scene_pass(true);   // final feat -> actA (bf16, cols 0..31)

    // ---- pixel MLP: 32->256->256->256->256->3 ----
    // Prefetch everything whose LDS slot is already dead: pix W1 frags (fragWs),
    // pix W2 frags (fragW2), pb1 (sb1), pb2 (sB4), pix W5 (sW1 slot).
    async_stage(OFF_FWS, pxF1, 16384,  tid);
    async_stage(OFF_FW2, pxF2, 131072, tid);
    async_stage(OFF_B1,  pb1,  1024,   tid);
    async_stage(OFF_B4,  pb2,  1024,   tid);
    async_stage(OFF_W1,  pW5,  3072,   tid);
    ASYNC_WAIT();
    __syncthreads();
    wave_gemm<1, 16, true, true, false>(actA, fragWs, sb1, actB, nullptr, wave, lane);
    __syncthreads();
    wave_gemm<8, 16, true, true, false>(actB, fragW2, sB4, actA, nullptr, wave, lane);
    __syncthreads();

    async_stage(OFF_FW2, pxF3, 131072, tid);
    async_stage(OFF_B4,  pb3,  1024,   tid);
    ASYNC_WAIT();
    __syncthreads();
    wave_gemm<8, 16, true, true, false>(actA, fragW2, sB4, actB, nullptr, wave, lane);
    __syncthreads();

    async_stage(OFF_FW2, pxF4, 131072, tid);
    async_stage(OFF_B4,  pb4,  1024,   tid);
    ASYNC_WAIT();
    __syncthreads();
    wave_gemm<8, 16, true, true, false>(actB, fragW2, sB4, actA, nullptr, wave, lane);
    __syncthreads();

    // final 256->3 in f32 scalar (pix_W5 already staged into sW1 slot)
    for (int idx = tid; idx < TILE * 3; idx += THREADS) {
        int p = idx / 3, cch = idx % 3;
        float acc = pb5[cch];
        for (int k = 0; k < 256; ++k)
            acc = fmaf((float)actA[p * ASTR + k], sW1[k * 3 + cch], acc);
        outRgb[((size_t)b * P_ + p0 + p) * 3 + cch] = acc;
    }
    if (tid < TILE) outD[(size_t)b * P_ + p0 + tid] = sD[tid];
}

// ---------------- launcher ----------------
extern "C" void kernel_launch(void* const* d_in, const int* in_sizes, int n_in,
                              void* d_out, int out_size, void* d_ws, size_t ws_size,
                              hipStream_t stream) {
    (void)in_sizes; (void)n_in; (void)out_size; (void)ws_size;
    const int*   id1 = (const int*)  d_in[0];
    const float* x   = (const float*)d_in[1];
    const float* R   = (const float*)d_in[2];
    const float* K   = (const float*)d_in[3];
    const float* T   = (const float*)d_in[4];
    const float* lat = (const float*)d_in[5];
    const float* hW1 = (const float*)d_in[6];
    const float* hb1 = (const float*)d_in[7];
    const float* hW2 = (const float*)d_in[8];
    const float* hb2 = (const float*)d_in[9];
    const float* Wih = (const float*)d_in[10];
    const float* Whh = (const float*)d_in[11];
    const float* bl  = (const float*)d_in[12];
    const float* Wd  = (const float*)d_in[13];
    const float* bd  = (const float*)d_in[14];
    const float* pW1 = (const float*)d_in[15];
    const float* pb1 = (const float*)d_in[16];
    const float* pW2 = (const float*)d_in[17];
    const float* pb2 = (const float*)d_in[18];
    const float* pW3 = (const float*)d_in[19];
    const float* pb3 = (const float*)d_in[20];
    const float* pW4 = (const float*)d_in[21];
    const float* pb4 = (const float*)d_in[22];
    const float* pW5 = (const float*)d_in[23];
    const float* pb5 = (const float*)d_in[24];
    const float* h0  = (const float*)d_in[25];
    const float* c0  = (const float*)d_in[26];

    char*   wsb    = (char*)d_ws;
    float*  theta  = (float*) (wsb + WS_THETA);
    float*  hidden = (float*) (wsb + WS_HIDDEN);
    __bf16* scF2   = (__bf16*)(wsb + WS_SCF2);
    __bf16* scF3   = (__bf16*)(wsb + WS_SCF3);
    __bf16* pxF1   = (__bf16*)(wsb + WS_PXF1);
    __bf16* pxF2   = (__bf16*)(wsb + WS_PXF2);
    __bf16* pxF3   = (__bf16*)(wsb + WS_PXF3);
    __bf16* pxF4   = (__bf16*)(wsb + WS_PXF4);

    (void)hipFuncSetAttribute(reinterpret_cast<const void*>(srn_raymarch),
                              hipFuncAttributeMaxDynamicSharedMemorySize, SMEM_BYTES);

    // hypernetwork -> theta
    hyper_hidden_kernel<<<dim3(B_), dim3(256), 0, stream>>>(id1, lat, hW1, hb1, hidden);
    hyper_theta_kernel<<<dim3((NSC + 255) / 256, B_), dim3(256), 0, stream>>>(hidden, hW2, hb2, theta);

    // one-time fragment packing (f32 -> bf16 WMMA-B images in workspace)
    for (int b = 0; b < B_; ++b) {
        pack_frags_kernel<<<dim3(256), dim3(256), 0, stream>>>(
            theta + (size_t)b * NSC + TH_W2, scF2 + (size_t)b * 256 * 256, 256, 256);
        pack_frags_kernel<<<dim3(32), dim3(256), 0, stream>>>(
            theta + (size_t)b * NSC + TH_W3, scF3 + (size_t)b * 256 * 32, 256, 32);
    }
    pack_frags_kernel<<<dim3(32),  dim3(256), 0, stream>>>(pW1, pxF1, 32,  256);
    pack_frags_kernel<<<dim3(256), dim3(256), 0, stream>>>(pW2, pxF2, 256, 256);
    pack_frags_kernel<<<dim3(256), dim3(256), 0, stream>>>(pW3, pxF3, 256, 256);
    pack_frags_kernel<<<dim3(256), dim3(256), 0, stream>>>(pW4, pxF4, 256, 256);

    float* outRgb = (float*)d_out;
    float* outD   = outRgb + (size_t)B_ * P_ * 3;
    srn_raymarch<<<dim3(P_ / TILE, B_), dim3(THREADS), SMEM_BYTES, stream>>>(
        x, R, K, T, Wih, Whh, bl, Wd, bd,
        pb1, pb2, pb3, pb4, pW5, pb5,
        h0, c0, theta, scF2, scF3, pxF1, pxF2, pxF3, pxF4,
        outRgb, outD);
}
